// RetentionHead_62311385530843
// MI455X (gfx1250) — compile-verified, matrix operand order checked by reading, not run on hardware
//
#include <hip/hip_runtime.h>
#include <hip/hip_bf16.h>

// ---------------------------------------------------------------------------
// RetNet retention head for MI455X (gfx1250, wave32, WMMA).
//   B=4, T=4096, C=1024, H=128, gamma = 0.96875, scale = C^-0.5 = 1/32
// Matrix-core bound (~47 GFLOP vs ~72MB) => all GEMMs on
// v_wmma_f32_16x16x32_bf16.  bf16 data paths, V kept transposed in global so
// every fragment is aligned b128 loads; K/V tiles double-buffered into LDS
// with async global->LDS copies (ASYNCcnt); causal split-K over blockIdx.y
// with f32 atomic commit for machine fill.
// ---------------------------------------------------------------------------

#define BDIM  4
#define TDIM  4096
#define CDIM  1024
#define HDIM  128
#define BT    (BDIM * TDIM)            // 16384 flat rows
#define KSLICES 4

#define LOG2_GAMMA  (-0.04580368961312479f)   // log2(0.96875)
#define QK_SCALE    (0.03125f)                // 1024^-0.5

typedef __attribute__((ext_vector_type(16))) __bf16 v16bf;
typedef __attribute__((ext_vector_type(8)))  float  v8f;

__device__ __forceinline__ v16bf make_frag(uint4 lo, uint4 hi) {
    union { uint4 u[2]; v16bf v; } f;
    f.u[0] = lo; f.u[1] = hi;
    return f.v;
}

__device__ __forceinline__ v8f wmma_bf16(v16bf a, v16bf b, v8f c) {
    return __builtin_amdgcn_wmma_f32_16x16x32_bf16(false, a, false, b,
                                                   (short)0, c, false, false);
}

// ---------------------------------------------------------------------------
// Prep: transpose+convert weights into wt[col][k] bf16 (768KB, L2-resident).
// ---------------------------------------------------------------------------
__global__ __launch_bounds__(128)
void prep_w_kernel(const float* __restrict__ Wq, const float* __restrict__ Wk,
                   const float* __restrict__ Wv, __bf16* __restrict__ wt) {
    const float* Wp[3] = {Wq, Wk, Wv};
    const float* W = Wp[blockIdx.x];
    const int col = threadIdx.x;                 // 0..127
    const int k0  = blockIdx.y * 128;
    const size_t dstbase = (size_t)(blockIdx.x * 128 + col) * CDIM;
    #pragma unroll 4
    for (int k = k0; k < k0 + 128; k += 8) {
        union { uint4 u; __bf16 h[8]; } p;
        #pragma unroll
        for (int j = 0; j < 8; ++j)
            p.h[j] = (__bf16)W[(size_t)(k + j) * HDIM + col];
        ((uint4*)wt)[(dstbase + k) >> 3] = p.u;
    }
}

// ---------------------------------------------------------------------------
// Projection: q,k row-major bf16; v TRANSPOSED (vt[h][row]).  2-wave blocks,
// 12 output tiles per wave.  All 12 B-fragments batch-loaded per k-step so
// the WMMA chain runs on partial loadcnt waits instead of load->wait->mma.
// ---------------------------------------------------------------------------
__global__ __launch_bounds__(64)
void retnet_proj_kernel(const float* __restrict__ x,
                        const __bf16* __restrict__ wt,
                        __bf16* __restrict__ qb,
                        __bf16* __restrict__ kb,
                        __bf16* __restrict__ vt) {
    const int lane = threadIdx.x & 31;
    const int wave = threadIdx.x >> 5;
    const int row0 = blockIdx.x * 16;
    const int M    = lane & 15;
    const int half = lane >> 4;

    v8f acc[12] = {};
    const float* xrow = x + (size_t)(row0 + M) * CDIM;

    for (int ks = 0; ks < 32; ++ks) {
        const int kc = ks * 32;
        // A fragment: two 8-float spans per half, cvt to bf16 in regs
        const float4* xr = (const float4*)(xrow + kc);
        float4 f0 = xr[half * 2 + 0], f1 = xr[half * 2 + 1];
        float4 f2 = xr[half * 2 + 4], f3 = xr[half * 2 + 5];
        v16bf a;
        a[0]=(__bf16)f0.x;  a[1]=(__bf16)f0.y;  a[2]=(__bf16)f0.z;  a[3]=(__bf16)f0.w;
        a[4]=(__bf16)f1.x;  a[5]=(__bf16)f1.y;  a[6]=(__bf16)f1.z;  a[7]=(__bf16)f1.w;
        a[8]=(__bf16)f2.x;  a[9]=(__bf16)f2.y;  a[10]=(__bf16)f2.z; a[11]=(__bf16)f2.w;
        a[12]=(__bf16)f3.x; a[13]=(__bf16)f3.y; a[14]=(__bf16)f3.z; a[15]=(__bf16)f3.w;

        // batch all 12 B-fragments, then 12 WMMA
        uint4 bl[12], bh[12];
        #pragma unroll
        for (int t = 0; t < 12; ++t) {
            const uint4* wr = (const uint4*)wt +
                (size_t)(wave * 192 + t * 16 + M) * 128 + ks * 4;
            bl[t] = wr[half];
            bh[t] = wr[half + 2];
        }
        #pragma unroll
        for (int t = 0; t < 12; ++t)
            acc[t] = wmma_bf16(a, make_frag(bl[t], bh[t]), acc[t]);

        __builtin_prefetch(xrow + kc + 64, 0, 3);   // stream-ahead on x
    }

    // epilogue
    #pragma unroll
    for (int t = 0; t < 12; ++t) {
        int ct = wave * 192 + t * 16;
        if (ct < 256) {                       // q,k: row-major bf16
            __bf16* dst = (ct < 128) ? qb : kb;
            int h0 = ct & 127;
            #pragma unroll
            for (int r = 0; r < 8; ++r)
                dst[(size_t)(row0 + r + 8 * half) * HDIM + h0 + M] =
                    (__bf16)acc[t][r];
        } else {                              // v: transposed, packed 16B store
            int h0 = ct - 256;
            union { uint4 u; __bf16 h[8]; } p;
            #pragma unroll
            for (int r = 0; r < 8; ++r) p.h[r] = (__bf16)acc[t][r];
            ((uint4*)vt)[(size_t)(h0 + M) * (BT / 8) + (row0 >> 3) + half] = p.u;
        }
    }
}

// ---------------------------------------------------------------------------
// Retention: block = 4 waves = 128 query rows (32/wave); blockIdx.y = key
// slice (stride-KSLICES interleave of 32-key blocks => balanced causal work).
// Per key block the workgroup shares one K tile + one Vt tile staged by
// async global->LDS b128 (double buffered, s_wait_asynccnt 8 overlap).
// Commit with f32 atomic adds (out pre-zeroed).
// ---------------------------------------------------------------------------
__global__ __launch_bounds__(128)
void retnet_attn_kernel(const __bf16* __restrict__ qb,
                        const __bf16* __restrict__ kb,
                        const __bf16* __restrict__ vt,
                        float* __restrict__ out) {
    __shared__ __align__(16) __bf16 kbuf[2][32 * 128];  // [m][h]   8KB each
    __shared__ __align__(16) __bf16 vbuf[2][128 * 32];  // [h][m]   8KB each
    __shared__ __align__(16) __bf16 pst[4][32 * 40];    // per-wave P relayout

    const int tid  = threadIdx.x;
    const int lane = tid & 31;
    const int wave = tid >> 5;
    const int slice = blockIdx.y;

    const int r0b     = blockIdx.x * 128;   // flat first query row of block
    const int batch   = r0b / TDIM;
    const int t0b     = r0b % TDIM;
    const int keyBase = batch * TDIM;
    const int r0w     = r0b + wave * 32;
    const int t0w     = t0b + wave * 32;

    const int M    = lane & 15;
    const int half = lane >> 4;

    // preload Q fragments (2 row-tiles x 4 K-steps over H)
    v16bf qf[2][4];
    #pragma unroll
    for (int rt = 0; rt < 2; ++rt) {
        const uint4* rp = (const uint4*)qb + (size_t)(r0w + rt * 16 + M) * 16;
        #pragma unroll
        for (int hs = 0; hs < 4; ++hs)
            qf[rt][hs] = make_frag(rp[hs * 4 + half], rp[hs * 4 + half + 2]);
    }

    v8f oacc[2][8] = {};
    const int nk = t0b / 32 + 4;                      // total key blocks
    const int nj = (nk > slice) ? (nk - slice + KSLICES - 1) / KSLICES : 0;

    auto stage = [&](int bi, int blk) {
        const int m0 = keyBase + blk * 32;
        #pragma unroll
        for (int i = 0; i < 4; ++i) {       // K tile: 512 16B chunks
            int c = i * 128 + tid;
            int m = c >> 4, hseg = c & 15;
            unsigned dst = (unsigned)(uintptr_t)&kbuf[bi][m * 128 + hseg * 8];
            unsigned long long src = (unsigned long long)(uintptr_t)
                (kb + (size_t)(m0 + m) * HDIM + hseg * 8);
            asm volatile("global_load_async_to_lds_b128 %0, %1, off"
                         :: "v"(dst), "v"(src) : "memory");
        }
        #pragma unroll
        for (int i = 0; i < 4; ++i) {       // Vt tile: 512 16B chunks
            int c = i * 128 + tid;
            int h = c >> 2, mseg = c & 3;
            unsigned dst = (unsigned)(uintptr_t)&vbuf[bi][h * 32 + mseg * 8];
            unsigned long long src = (unsigned long long)(uintptr_t)
                (vt + (size_t)h * BT + m0 + mseg * 8);
            asm volatile("global_load_async_to_lds_b128 %0, %1, off"
                         :: "v"(dst), "v"(src) : "memory");
        }
    };

    if (nj > 0) stage(0, slice);
    for (int j = 0; j < nj; ++j) {
        const int bi = j & 1;
        const bool more = (j + 1) < nj;
        if (more) {
            stage(bi ^ 1, slice + (j + 1) * KSLICES);
            asm volatile("s_wait_asynccnt 0x8" ::: "memory"); // cur tile done
        } else {
            asm volatile("s_wait_asynccnt 0x0" ::: "memory");
        }
        __syncthreads();

        const int mt0 = (slice + j * KSLICES) * 32;

        // ---- wei = Q K^T ----
        v8f sw[2][2] = {};
        const uint4* kv = (const uint4*)kbuf[bi];
        #pragma unroll
        for (int hs = 0; hs < 4; ++hs) {
            #pragma unroll
            for (int mh = 0; mh < 2; ++mh) {
                int m = mh * 16 + M;
                v16bf kf = make_frag(kv[m * 16 + hs * 4 + half],
                                     kv[m * 16 + hs * 4 + half + 2]);
                #pragma unroll
                for (int rt = 0; rt < 2; ++rt)
                    sw[rt][mh] = wmma_bf16(qf[rt][hs], kf, sw[rt][mh]);
            }
        }

        // ---- decay + scale, C-layout -> row-major P in LDS ----
        #pragma unroll
        for (int rt = 0; rt < 2; ++rt)
            #pragma unroll
            for (int mh = 0; mh < 2; ++mh)
                #pragma unroll
                for (int r = 0; r < 8; ++r) {
                    int n_t = t0w + rt * 16 + r + 8 * half;
                    int m_t = mt0 + mh * 16 + M;
                    int d   = n_t - m_t;
                    float f = (d >= 0)
                        ? QK_SCALE * __builtin_amdgcn_exp2f((float)d * LOG2_GAMMA)
                        : 0.0f;
                    pst[wave][(rt * 16 + r + 8 * half) * 40 + mh * 16 + M] =
                        (__bf16)(sw[rt][mh][r] * f);
                }
        asm volatile("" ::: "memory");      // LDS in-order within wave

        // ---- out += P V ----
        v16bf pa[2];
        const uint4* pv = (const uint4*)pst[wave];
        #pragma unroll
        for (int rt = 0; rt < 2; ++rt)
            pa[rt] = make_frag(pv[(rt * 16 + M) * 5 + half],
                               pv[(rt * 16 + M) * 5 + half + 2]);
        const uint4* vv = (const uint4*)vbuf[bi];
        #pragma unroll
        for (int ht = 0; ht < 8; ++ht) {
            int h = ht * 16 + M;
            v16bf vf = make_frag(vv[h * 4 + half], vv[h * 4 + half + 2]);
            #pragma unroll
            for (int rt = 0; rt < 2; ++rt)
                oacc[rt][ht] = wmma_bf16(pa[rt], vf, oacc[rt][ht]);
        }
        __syncthreads();   // tile bi free for restaging next iteration
    }

    // ---- commit partial sums: f32 atomic add (out pre-zeroed) ----
    #pragma unroll
    for (int rt = 0; rt < 2; ++rt)
        #pragma unroll
        for (int ht = 0; ht < 8; ++ht)
            #pragma unroll
            for (int r = 0; r < 8; ++r) {
                size_t idx = (size_t)(r0w + rt * 16 + r + 8 * half) * HDIM
                             + ht * 16 + M;
                __hip_atomic_fetch_add(&out[idx], oacc[rt][ht][r],
                                       __ATOMIC_RELAXED,
                                       __HIP_MEMORY_SCOPE_AGENT);
            }
}

// ---------------------------------------------------------------------------
extern "C" void kernel_launch(void* const* d_in, const int* in_sizes, int n_in,
                              void* d_out, int out_size, void* d_ws, size_t ws_size,
                              hipStream_t stream) {
    const float* x  = (const float*)d_in[0];
    const float* Wq = (const float*)d_in[1];
    const float* Wk = (const float*)d_in[2];
    const float* Wv = (const float*)d_in[3];
    float* out = (float*)d_out;

    // workspace: q, k row-major; vt transposed; wt transposed weights (bf16)
    __bf16* qb = (__bf16*)d_ws;
    __bf16* kb = qb + (size_t)BT * HDIM;
    __bf16* vt = kb + (size_t)BT * HDIM;
    __bf16* wt = vt + (size_t)BT * HDIM;          // 384*1024 bf16

    hipMemsetAsync(out, 0, (size_t)out_size * sizeof(float), stream);
    prep_w_kernel<<<dim3(3, 8), 128, 0, stream>>>(Wq, Wk, Wv, wt);
    retnet_proj_kernel<<<BT / 16, 64, 0, stream>>>(x, wt, qb, kb, vt);
    retnet_attn_kernel<<<dim3(BT / 128, KSLICES), 128, 0, stream>>>(qb, kb, vt, out);
}
